// ExpressionPerformer_10213432230563
// MI455X (gfx1250) — matrix operand order repaired and checked by reading, not compiled, not run on hardware
//
#include <hip/hip_runtime.h>
#include <hip/hip_bf16.h>
#include <math.h>

typedef __bf16 bf16_t;
typedef __attribute__((ext_vector_type(16))) __bf16 v16bf;
typedef __attribute__((ext_vector_type(8)))  __bf16 v8bf;
typedef __attribute__((ext_vector_type(8)))  float  v8f;

namespace cfg {
constexpr int B = 16, G = 2048, D = 256, H = 8, FFN = 1024, L = 4;
constexpr int BG = B * G;   // 32768 token rows
constexpr int DH = D / H;   // 32
}

// low 32 bits of a flat LDS pointer == LDS byte offset (ISA 10.2)
__device__ __forceinline__ unsigned lds_addr32(const void* p) {
  return (unsigned)(unsigned long long)p;
}

// ---------------- weight transpose + bf16 convert (folded over layers) ----------------
__global__ void wtrans_kernel(const float* __restrict__ W, bf16_t* __restrict__ WT,
                              int R, int C, size_t total) {
  size_t idx = (size_t)blockIdx.x * blockDim.x + threadIdx.x;
  if (idx >= total) return;
  size_t rc = (size_t)R * C;
  size_t l = idx / rc, rem = idx - l * rc;
  int r = (int)(rem / C), c = (int)(rem - (size_t)r * C);
  WT[l * rc + (size_t)c * R + r] = (bf16_t)W[idx];
}

// ---------------- rotary expression embedding ----------------
__global__ void embed_kernel(const float* __restrict__ x, const float* __restrict__ gene_emb,
                             const float* __restrict__ inv_freq, float* __restrict__ h) {
  using namespace cfg;
  size_t idx = (size_t)blockIdx.x * blockDim.x + threadIdx.x;   // over BG*D
  int d = (int)(idx & (D - 1));
  size_t bg = idx >> 8;
  int g = (int)(bg & (G - 1));
  float xv = x[bg];
  float e = (d < D / 2) ? sinf(xv * inv_freq[d]) : cosf(xv * inv_freq[d - D / 2]);
  if (xv == -10.0f) e = 0.0f;
  h[idx] = gene_emb[(size_t)g * D + d] + e;
}

// ---------------- layernorm -> bf16 (one wave per row of 256) ----------------
__global__ __launch_bounds__(256)
void ln_bf16_kernel(const float* __restrict__ h, const float* __restrict__ gam,
                    const float* __restrict__ bet, bf16_t* __restrict__ out) {
  using namespace cfg;
  const int lane = threadIdx.x & 31;
  const size_t row = (size_t)blockIdx.x * 8 + (threadIdx.x >> 5);
  const float* hp = h + row * D;
  float xv[8], s = 0.f, sq = 0.f;
#pragma unroll
  for (int i = 0; i < 8; ++i) {
    xv[i] = hp[lane + 32 * i];
    s += xv[i]; sq += xv[i] * xv[i];
  }
#pragma unroll
  for (int o = 16; o; o >>= 1) { s += __shfl_xor(s, o); sq += __shfl_xor(sq, o); }
  float m = s * (1.0f / 256.0f);
  float var = sq * (1.0f / 256.0f) - m * m;
  float rstd = rsqrtf(var + 1e-5f);
  bf16_t* op = out + row * D;
#pragma unroll
  for (int i = 0; i < 8; ++i) {
    int d = lane + 32 * i;
    op[d] = (bf16_t)((xv[i] - m) * rstd * gam[d] + bet[d]);
  }
}

// ---------------- WMMA bf16 GEMM with async-LDS staged weight tiles ----------------
// C = act(A[MxK] @ WT^T + bias); WT is [N x K] row-major (pre-transposed).
// Per block: 8 waves x 16 rows = 128 rows, 64-col N slab. The 64x32 bf16 W tile
// (4KB) is DMA'd to LDS once per k-step via global_load_async_to_lds_b128
// (double-buffered, overlapped with WMMA). All 4 B-fragments are pulled from
// LDS up-front so the scheduler can use partial DScnt waits across the WMMAs.
#define MODE_F32       0
#define MODE_F32_SQ    1
#define MODE_BF16_GELU 2
#define MODE_F32_RES   3

__device__ __forceinline__ v16bf load_a_frag(const bf16_t* __restrict__ A, int ldk,
                                             int row, int k0, int lane) {
  // A 16x32 bf16 fragment: lanes 0-15 -> K {0..15}, lanes 16-31 -> K {16..31}
  const int koff = (lane >> 4) * 16;
  const bf16_t* p = A + (size_t)row * ldk + k0 + koff;
  v8bf lo = *(const v8bf*)p;
  v8bf hi = *(const v8bf*)(p + 8);
  v16bf a;
#pragma unroll
  for (int i = 0; i < 8; ++i) { a[i] = lo[i]; a[i + 8] = hi[i]; }
  return a;
}

template <int MODE>
__global__ __launch_bounds__(256)
void gemm_wmma_kernel(const bf16_t* __restrict__ A, const bf16_t* __restrict__ WT,
                      const float* __restrict__ bias, void* __restrict__ Cout,
                      const float* __restrict__ resid, int N, int K) {
  __shared__ __align__(32) bf16_t btile[2][64 * 32];   // [buf][n][k] 4KB each

  const int tid  = threadIdx.x;
  const int lane = tid & 31;
  const int wave = tid >> 5;
  const int m0 = (blockIdx.x * 8 + wave) * 16;         // 16-row strip per wave
  const int n_slab = blockIdx.y * 64;                  // 4 n-tiles per wave
  const int row = m0 + (lane & 15);
  const int bkoff = (lane >> 4) * 16;                  // B frag: lanes16-31 K+=16

  // staging: each of the 256 threads DMAs one 16B chunk: n = tid>>2, quarter = tid&3
  const int sn = tid >> 2, sq = tid & 3;
  const bf16_t* gsrc = WT + (size_t)(n_slab + sn) * K + sq * 8;
  const unsigned lds_dst0 = lds_addr32(&btile[0][sn * 32 + sq * 8]);
  const unsigned lds_dst1 = lds_addr32(&btile[1][sn * 32 + sq * 8]);

  v8f acc[4];
#pragma unroll
  for (int i = 0; i < 4; ++i)
#pragma unroll
    for (int j = 0; j < 8; ++j) acc[i][j] = 0.0f;

  const int ksteps = K >> 5;

  // per-k-step compute body: preload all 4 B-fragments, then 4 WMMAs
  auto compute = [&](int ks, int buf) {
    v16bf a = load_a_frag(A, K, row, ks << 5, lane);
    v16bf bfr[4];
#pragma unroll
    for (int nt = 0; nt < 4; ++nt) {
      const int nl = nt * 16 + (lane & 15);
      bfr[nt] = *(const v16bf*)&btile[buf][nl * 32 + bkoff];
    }
#pragma unroll
    for (int nt = 0; nt < 4; ++nt)
      acc[nt] = __builtin_amdgcn_wmma_f32_16x16x32_bf16(
          false, a, false, bfr[nt], (short)0, acc[nt], false, false);
  };

  // prologue: stage k-step 0 into buffer 0
  {
    unsigned long long g = (unsigned long long)gsrc;
    asm volatile("global_load_async_to_lds_b128 %0, %1, off"
                 :: "v"(lds_dst0), "v"(g) : "memory");
  }

  // steady state: branch-free, always stage-next + wait<=1
  for (int ks = 0; ks < ksteps - 1; ++ks) {
    const int buf = ks & 1;
    {
      unsigned long long g = (unsigned long long)(gsrc + (ks + 1) * 32);
      asm volatile("global_load_async_to_lds_b128 %0, %1, off"
                   :: "v"(buf ? lds_dst0 : lds_dst1), "v"(g) : "memory");
    }
    asm volatile("s_wait_asynccnt 0x1" ::: "memory");   // current tile's DMA done
    __syncthreads();                                    // tile visible block-wide
    compute(ks, buf);
    __syncthreads();                                    // done reading before overwrite
  }

  // epilogue: last k-step
  asm volatile("s_wait_asynccnt 0x0" ::: "memory");
  __syncthreads();
  compute(ksteps - 1, (ksteps - 1) & 1);

#pragma unroll
  for (int nt = 0; nt < 4; ++nt) {
    const int col = n_slab + nt * 16 + (lane & 15);
    const float bcol = bias[col];
#pragma unroll
    for (int r = 0; r < 8; ++r) {
      const int rr = m0 + ((lane >> 4) * 8) + r;        // C/D: M = r + 8*(lane>=16)
      const size_t idx = (size_t)rr * N + col;
      float v = acc[nt][r] + bcol;
      if (MODE == MODE_F32) {
        ((float*)Cout)[idx] = v;
      } else if (MODE == MODE_F32_SQ) {
        ((float*)Cout)[idx] = v * v;                    // feature map 'sqr'
      } else if (MODE == MODE_BF16_GELU) {
        float gl = 0.5f * v * (1.0f + erff(v * 0.70710678118654752f));
        ((bf16_t*)Cout)[idx] = (bf16_t)gl;
      } else {                                          // MODE_F32_RES
        ((float*)Cout)[idx] = v + resid[idx];
      }
    }
  }
}

// ---------------- causal linear attention: one wave per (b, head) ----------------
__global__ __launch_bounds__(256)
void attn_kernel(const float* __restrict__ qb, const float* __restrict__ kb,
                 const float* __restrict__ vb, float* __restrict__ h) {
  using namespace cfg;
  const int lane = threadIdx.x & 31;                 // lane == output dim d
  const int pair = blockIdx.x * 8 + (threadIdx.x >> 5);
  const int b  = pair >> 3;
  const int hh = pair & 7;
  const size_t base = (size_t)b * G * D + (size_t)hh * DH + lane;

  float num[32];                                     // num_state[f][d=lane]
#pragma unroll
  for (int f = 0; f < 32; ++f) num[f] = 0.0f;
  float dens = 0.0f;                                 // den_state[f=lane]

  for (int t = 0; t < G; ++t) {
    const size_t off = base + (size_t)t * D;
    const float qv = qb[off], kv = kb[off], vv = vb[off];
    float acc = 0.0f;
#pragma unroll
    for (int f = 0; f < 32; ++f) {                   // wave32 broadcast via readlane
      float kf = __int_as_float(__builtin_amdgcn_readlane(__float_as_int(kv), f));
      float qf = __int_as_float(__builtin_amdgcn_readlane(__float_as_int(qv), f));
      num[f] += kf * vv;
      acc += qf * num[f];
    }
    dens += kv;
    float part = qv * dens;
#pragma unroll
    for (int o = 16; o; o >>= 1) part += __shfl_xor(part, o);
    h[off] += acc / (part + 1e-16f);
  }
}

// ---------------- output head: out = h @ Wout + bout ----------------
__global__ __launch_bounds__(256)
void head_kernel(const float* __restrict__ h, const float* __restrict__ Wout,
                 const float* __restrict__ bout, float* __restrict__ out) {
  using namespace cfg;
  const int lane = threadIdx.x & 31;
  const size_t row = (size_t)blockIdx.x * 8 + (threadIdx.x >> 5);
  const float* hp = h + row * D;
  float s = 0.0f;
#pragma unroll
  for (int i = 0; i < 8; ++i) {
    int d = lane + 32 * i;
    s += hp[d] * Wout[d];
  }
#pragma unroll
  for (int o = 16; o; o >>= 1) s += __shfl_xor(s, o);
  if (lane == 0) out[row] = s + bout[0];
}

// ---------------- host orchestration ----------------
extern "C" void kernel_launch(void* const* d_in, const int* in_sizes, int n_in,
                              void* d_out, int out_size, void* d_ws, size_t ws_size,
                              hipStream_t stream) {
  using namespace cfg;
  const float* x        = (const float*)d_in[0];
  const float* gene_emb = (const float*)d_in[1];
  const float* inv_freq = (const float*)d_in[2];
  const float* Wq   = (const float*)d_in[3];
  const float* bq   = (const float*)d_in[4];
  const float* Wk   = (const float*)d_in[5];
  const float* bk   = (const float*)d_in[6];
  const float* Wv   = (const float*)d_in[7];
  const float* bv   = (const float*)d_in[8];
  const float* ln1g = (const float*)d_in[9];
  const float* ln1b = (const float*)d_in[10];
  const float* ln2g = (const float*)d_in[11];
  const float* ln2b = (const float*)d_in[12];
  const float* WU   = (const float*)d_in[13];
  const float* bU   = (const float*)d_in[14];
  const float* WV   = (const float*)d_in[15];
  const float* bV   = (const float*)d_in[16];
  const float* Wout = (const float*)d_in[17];
  const float* bout = (const float*)d_in[18];
  float* out = (float*)d_out;

  // workspace carve-up (256B aligned slices)
  char* ws = (char*)d_ws;
  size_t off = 0;
  auto take = [&](size_t bytes) { void* p = ws + off; off += (bytes + 255) & ~(size_t)255; return p; };
  float*  h   = (float*) take((size_t)BG * D * 4);
  bf16_t* hn  = (bf16_t*)take((size_t)BG * D * 2);
  float*  qb2 = (float*) take((size_t)BG * D * 4);
  float*  kb2 = (float*) take((size_t)BG * D * 4);
  float*  vb2 = (float*) take((size_t)BG * D * 4);
  bf16_t* Ub  = (bf16_t*)take((size_t)BG * FFN * 2);
  bf16_t* WqT = (bf16_t*)take((size_t)L * D * D * 2);
  bf16_t* WkT = (bf16_t*)take((size_t)L * D * D * 2);
  bf16_t* WvT = (bf16_t*)take((size_t)L * D * D * 2);
  bf16_t* WUT = (bf16_t*)take((size_t)L * D * FFN * 2);
  bf16_t* WVT = (bf16_t*)take((size_t)L * FFN * D * 2);
  (void)ws_size; (void)in_sizes; (void)n_in; (void)out_size;

  // weights -> transposed bf16 (stay L2-resident)
  {
    size_t tQ = (size_t)L * D * D;
    wtrans_kernel<<<(tQ + 255) / 256, 256, 0, stream>>>(Wq, WqT, D, D, tQ);
    wtrans_kernel<<<(tQ + 255) / 256, 256, 0, stream>>>(Wk, WkT, D, D, tQ);
    wtrans_kernel<<<(tQ + 255) / 256, 256, 0, stream>>>(Wv, WvT, D, D, tQ);
    size_t tU = (size_t)L * D * FFN;
    wtrans_kernel<<<(tU + 255) / 256, 256, 0, stream>>>(WU, WUT, D, FFN, tU);
    wtrans_kernel<<<(tU + 255) / 256, 256, 0, stream>>>(WV, WVT, FFN, D, tU);
  }

  // embedding
  embed_kernel<<<(size_t)BG * D / 256, 256, 0, stream>>>(x, gene_emb, inv_freq, h);

  const dim3 blk(256);
  const dim3 grid_qkv(BG / 128, D / 64);     // (256, 4)
  const dim3 grid_ffn1(BG / 128, FFN / 64);  // (256, 16)
  const dim3 grid_ffn2(BG / 128, D / 64);    // (256, 4)

  for (int l = 0; l < L; ++l) {
    ln_bf16_kernel<<<BG / 8, blk, 0, stream>>>(h, ln1g + l * D, ln1b + l * D, hn);

    gemm_wmma_kernel<MODE_F32_SQ><<<grid_qkv, blk, 0, stream>>>(
        hn, WqT + (size_t)l * D * D, bq + l * D, qb2, nullptr, D, D);
    gemm_wmma_kernel<MODE_F32_SQ><<<grid_qkv, blk, 0, stream>>>(
        hn, WkT + (size_t)l * D * D, bk + l * D, kb2, nullptr, D, D);
    gemm_wmma_kernel<MODE_F32><<<grid_qkv, blk, 0, stream>>>(
        hn, WvT + (size_t)l * D * D, bv + l * D, vb2, nullptr, D, D);

    attn_kernel<<<B * H / 8, blk, 0, stream>>>(qb2, kb2, vb2, h);

    ln_bf16_kernel<<<BG / 8, blk, 0, stream>>>(h, ln2g + l * D, ln2b + l * D, hn);

    gemm_wmma_kernel<MODE_BF16_GELU><<<grid_ffn1, blk, 0, stream>>>(
        hn, WUT + (size_t)l * D * FFN, bU + l * FFN, Ub, nullptr, FFN, D);
    gemm_wmma_kernel<MODE_F32_RES><<<grid_ffn2, blk, 0, stream>>>(
        Ub, WVT + (size_t)l * FFN * D, bV + l * D, h, h, D, FFN);
  }

  head_kernel<<<BG / 8, blk, 0, stream>>>(h, Wout, bout, out);
}
